// C3_22720376995990
// MI455X (gfx1250) — matrix operand order repaired and checked by reading, not compiled, not run on hardware
//
#include <hip/hip_runtime.h>

typedef __attribute__((ext_vector_type(16))) _Float16 v16h;
typedef __attribute__((ext_vector_type(4)))  _Float16 v4h;
typedef __attribute__((ext_vector_type(8)))  float    v8f;
typedef int v4i_vs __attribute__((vector_size(16)));   // matches async-LDS builtin param

#define OC      16
#define NCH     6
#define NKC     8            // 8 chunks of K=32 (padded K = 256, 30 groups of 8 used)
#define NGRP    30           // valid (ky,c) groups: 5*6
#define TY      16           // output rows per workgroup tile
#define TX      128          // output cols per workgroup tile
#define IN_H    256
#define IN_W    256
#define OUT_HW  252
#define LW      136          // halves per (y,c) row in LDS (132 used + zero pad)
#define YSTR    (NCH * LW)   // 816 halves per y row
#define S_VALID (20 * YSTR)  // 16320 halves
#define S_TOTAL 16608        // + 288-half zeroed tail for phantom groups 30,31
#define A_BYTES (NKC * 512 * 2)   // 8192 B of packed A fragments

#define AS1 __attribute__((address_space(1)))
#define AS3 __attribute__((address_space(3)))

#if defined(__AMDGCN__) && __has_builtin(__builtin_amdgcn_global_load_async_to_lds_b128)
#define HAVE_ASYNC_LDS 1
#else
#define HAVE_ASYNC_LDS 0
#endif

__constant__ int d_FM[16][6] = {
  {0,1,2,0,0,0},{1,2,3,0,0,0},{2,3,4,0,0,0},{3,4,5,0,0,0},{0,4,5,0,0,0},{0,1,5,0,0,0},
  {0,1,2,3,0,0},{1,2,3,4,0,0},{2,3,4,5,0,0},{0,3,4,5,0,0},{0,1,4,5,0,0},{0,1,2,5,0,0},
  {0,1,3,4,0,0},{1,2,4,5,0,0},{0,2,3,5,0,0},
  {0,1,2,3,4,5}};
__constant__ int d_FI[16] = {3,3,3,3,3,3,4,4,4,4,4,4,4,4,4,6};

// ---------------------------------------------------------------------------
// Kernel 1: binarize weights, expand to dense 16x(6ch x 25tap), and pack in
// WMMA A-fragment order with the padded-K enumeration:
//   chunk slot s (0..31)  ->  group g = kc*4 + s/8 (= ky*6 + c), tap kx = s%8
//   A fragment: row M = lane&15, lo = lane>>4, K_local(h) = lo*8 + h + (h>=8?8:0)
// wsA[kc*512 + lane*16 + h]: per-lane contiguous 32B block per chunk.
// ---------------------------------------------------------------------------
__global__ void binpack_kernel(const float* __restrict__ W3,
                               const float* __restrict__ W4,
                               const float* __restrict__ W6,
                               _Float16* __restrict__ wsA) {
  __shared__ float s_dense[16][152];
  const int tid = threadIdx.x;

  if (tid < 16) {
    const int o = tid;
    const int I = d_FI[o];
    const float* src = (o < 6) ? (W3 + o * 75)
                               : ((o < 15) ? (W4 + (o - 6) * 100) : W6);
    float asum = 0.f;
    for (int tap = 0; tap < 25; ++tap) {
      float m = 0.f;
      for (int i = 0; i < I; ++i) m += src[i * 25 + tap];
      m /= (float)I;
      for (int i = 0; i < I; ++i) {
        float v = src[i * 25 + tap] - m;
        v = fminf(1.f, fmaxf(-1.f, v));
        asum += fabsf(v);
      }
    }
    const float alpha = asum / (float)(I * 25);

    for (int k = 0; k < 152; ++k) s_dense[o][k] = 0.f;
    for (int i = 0; i < I; ++i) {
      const int c = d_FM[o][i];
      for (int tap = 0; tap < 25; ++tap) {
        float m = 0.f;
        for (int j = 0; j < I; ++j) m += src[j * 25 + tap];
        m /= (float)I;
        float v = src[i * 25 + tap] - m;
        v = fminf(1.f, fmaxf(-1.f, v));
        const float s = (v > 0.f) ? alpha : ((v < 0.f) ? -alpha : 0.f);
        s_dense[o][c * 25 + tap] = s;
      }
    }
  }
  __syncthreads();

  for (int idx = tid; idx < NKC * 32 * 16; idx += blockDim.x) {
    const int kc  = idx >> 9;
    const int rem = idx & 511;
    const int l   = rem >> 4;
    const int h   = rem & 15;
    const int lo  = l >> 4;
    const int M   = l & 15;
    const int kl  = lo * 8 + h + ((h >= 8) ? 8 : 0);   // chunk slot s = 0..31
    const int g   = kc * 4 + (kl >> 3);                // = ky*6 + c
    const int kx  = kl & 7;
    float v = 0.f;
    if (g < NGRP && kx < 5) {
      const int ky = g / 6;
      const int c  = g - ky * 6;
      v = s_dense[M][c * 25 + ky * 5 + kx];
    }
    wsA[idx] = (_Float16)v;
  }
}

// ---------------------------------------------------------------------------
// Kernel 2: implicit-GEMM 5x5 conv via v_wmma_f32_16x16x32_f16.
// - Weight matrix (8 KB) broadcast to LDS via global_load_async_to_lds_b128
//   (ASYNCcnt), overlapped with the f32->f16 input staging.
// - LDS layout s_in[y][c][x] (stride LW) makes the B address linear in the
//   group index g, so each 8-half run is one ds_load_b128 off ONE base VGPR
//   with an immediate offset.
// ---------------------------------------------------------------------------
__global__ __launch_bounds__(256) void conv_wmma_kernel(
    const float* __restrict__ in, const _Float16* __restrict__ wsA,
    float* __restrict__ out) {
  __shared__ _Float16 s_in[S_TOTAL];
  __shared__ _Float16 s_A[NKC * 512];

  const int tid  = threadIdx.x;
  const int lane = tid & 31;
  const int wv   = tid >> 5;
  const int lo   = lane >> 4;
  const int col  = lane & 15;
  const int x0   = blockIdx.x * TX;
  const int y0   = blockIdx.y * TY;
  const int n    = blockIdx.z;

  // ---- async DMA of the packed A matrix into LDS (no VGPR destinations) ----
#if HAVE_ASYNC_LDS
  {
    const char* gsrc = (const char*)wsA + tid * 16;   // 256 thr x 16 B = 4 KB
    char*       ldst = (char*)s_A + tid * 16;
    __builtin_amdgcn_global_load_async_to_lds_b128((AS1 v4i_vs*)gsrc,
                                                   (AS3 v4i_vs*)ldst, 0, 0);
    __builtin_amdgcn_global_load_async_to_lds_b128((AS1 v4i_vs*)gsrc,
                                                   (AS3 v4i_vs*)ldst, 4096, 0);
  }
#else
  for (int i = tid; i < A_BYTES / 8; i += 256)
    ((unsigned long long*)s_A)[i] = ((const unsigned long long*)wsA)[i];
#endif

  // ---- stage input tile f32 -> f16 into LDS, vectorized x4 (overlaps the
  // async weight DMA). Covers the whole padded array (x in [132,136) and the
  // 288-half tail) with zeros so every padding slot B touches is finite. ----
  const float* inb = in + (size_t)n * NCH * IN_H * IN_W;
  for (int i = tid; i < S_TOTAL / 4; i += 256) {
    const int e = i * 4;
    const int y = e / YSTR;
    const int r = e - y * YSTR;
    const int c = r / LW;
    const int x = r - c * LW;            // multiple of 4
    const int gy = y0 + y;
    const int gx = x0 + x;               // multiple of 4 -> float4 aligned
    v4h hv;
    hv[0] = (_Float16)0.f; hv[1] = (_Float16)0.f;
    hv[2] = (_Float16)0.f; hv[3] = (_Float16)0.f;
    if (y < 20 && x < 132 && gy < IN_H && gx < IN_W) {
      const float4 f = *(const float4*)(inb + (c * IN_H + gy) * IN_W + gx);
      hv[0] = (_Float16)f.x; hv[1] = (_Float16)f.y;
      hv[2] = (_Float16)f.z; hv[3] = (_Float16)f.w;
    }
    *(v4h*)(s_in + e) = hv;
  }

#if HAVE_ASYNC_LDS
#if __has_builtin(__builtin_amdgcn_s_wait_asynccnt)
  __builtin_amdgcn_s_wait_asynccnt(0);
#else
  asm volatile("s_wait_asynccnt 0x0" ::: "memory");
#endif
#endif
  __syncthreads();

  // A fragments from LDS: 8 x v16h (2 x ds_load_b128 each)
  v16h A[NKC];
  #pragma unroll
  for (int kc = 0; kc < NKC; ++kc)
    A[kc] = *(const v16h*)(s_A + kc * 512 + lane * 16);

  const int hb = lo * LW;                // lo-dependent part of the base

  for (int t = 0; t < 16; ++t) {
    const int ry = wv * 2 + (t >> 3);    // local output row
    const int sx = (t & 7) * 16;         // strip start col
    const int vb = ry * YSTR + sx + col + hb;   // one base address per strip
    v8f acc = {};
    #pragma unroll
    for (int kc = 0; kc < NKC; ++kc) {
      v16h b;
      #pragma unroll
      for (int j = 0; j < 2; ++j) {
        #pragma unroll
        for (int q = 0; q < 8; ++q) {
          b[j * 8 + q] = s_in[vb + (kc * 4 + 2 * j) * LW + q];  // imm offsets
        }
      }
      acc = __builtin_amdgcn_wmma_f32_16x16x32_f16(
          false, A[kc], false, b, (short)0, acc, false, false);
    }

    // D layout: VGPR r -> channel (r + 8*lo), lane&15 -> pixel col
    const int gy = y0 + ry;
    const int gx = x0 + sx + col;
    if (gy < OUT_HW && gx < OUT_HW) {
      #pragma unroll
      for (int r8 = 0; r8 < 8; ++r8) {
        const int ch = r8 + 8 * lo;
        out[(((size_t)n * OC + ch) * OUT_HW + gy) * OUT_HW + gx] = acc[r8];
      }
    }
  }
}

extern "C" void kernel_launch(void* const* d_in, const int* in_sizes, int n_in,
                              void* d_out, int out_size, void* d_ws, size_t ws_size,
                              hipStream_t stream) {
  const float* inp = (const float*)d_in[0];
  const float* W3  = (const float*)d_in[1];
  const float* W4  = (const float*)d_in[2];
  const float* W6  = (const float*)d_in[3];
  float* out = (float*)d_out;
  _Float16* wsA = (_Float16*)d_ws;   // 8*32*16 halves = 8192 B

  binpack_kernel<<<1, 256, 0, stream>>>(W3, W4, W6, wsA);

  dim3 grid((OUT_HW + TX - 1) / TX, (OUT_HW + TY - 1) / TY, 64);
  conv_wmma_kernel<<<grid, 256, 0, stream>>>(inp, wsA, out);
}
